// FeatureFusionModule_66597762891910
// MI455X (gfx1250) — matrix operand order repaired and checked by reading, not compiled, not run on hardware
//
#include <hip/hip_runtime.h>
#include <hip/hip_bf16.h>
#include <cstdint>
#include <cstddef>

#define BB 4
#define C1 256
#define C2 128
#define CC 256
#define NN 4096
#define BN_EPS 1e-5f

typedef __attribute__((ext_vector_type(16))) __bf16 v16bf;
typedef __attribute__((ext_vector_type(8)))  __bf16 v8bf;
typedef __attribute__((ext_vector_type(8)))  float  v8f;

// ---- CDNA5 async global->LDS copy (16B per lane) + counted waits ----------
__device__ __forceinline__ void async_copy_b128(const void* g, void* l) {
  unsigned lds_addr = (unsigned)(unsigned long long)l;
  asm volatile("global_load_async_to_lds_b128 %0, %1, off"
               :: "v"(lds_addr), "v"(g) : "memory");
}
__device__ __forceinline__ void wait_async0() {
  asm volatile("s_wait_asynccnt 0x0" ::: "memory");
}
// <=16 outstanding: drains everything older than the 16 just-issued
// prefetch instructions of the *next* tile, i.e. current tile is resident.
__device__ __forceinline__ void wait_async16() {
  asm volatile("s_wait_asynccnt 0x10" ::: "memory");
}

// A-fragment: contiguous k order of this code (B matches); per lane one 16-element contiguous run
// (K 0-15 for lanes 0-15; K 16-31 for lanes 16-31).
__device__ __forceinline__ v16bf make_afrag(const __bf16* ap) {
  v8bf lo = *(const v8bf*)ap;
  v8bf hi = *(const v8bf*)(ap + 8);
  return __builtin_shufflevector(lo, hi, 0, 1, 2, 3, 4, 5, 6, 7,
                                 8, 9, 10, 11, 12, 13, 14, 15);
}

// ===========================================================================
// Projection GEMM: out = W(C,CIN) * X(B,CIN,N) + bias, bf16 output.
// OUT_NC: out[b][n][c] (point-major)  else out[b][c][n] (channel-major).
// 64x64 output tile per workgroup, 8 waves, each wave two 16x16 WMMA tiles.
// ===========================================================================
template<int CIN, bool OUT_NC>
__global__ __launch_bounds__(256) void proj_kernel(
    const float* __restrict__ X, const float* __restrict__ W,
    const float* __restrict__ bias, __bf16* __restrict__ out)
{
  __shared__ __bf16 a_lds[64 * CIN];   // [n_local][ci]  (X transposed)
  __shared__ __bf16 b_lds[64 * CIN];   // [c_local][ci]  (W rows)
  const int b  = blockIdx.z;
  const int n0 = blockIdx.x * 64;
  const int c0 = blockIdx.y * 64;
  const int t  = threadIdx.x;

  for (int id = t; id < 64 * CIN; id += 256) {
    int ci = id >> 6, nl = id & 63;
    a_lds[nl * CIN + ci] = (__bf16)X[((size_t)b * CIN + ci) * NN + n0 + nl];
  }
  for (int id = t; id < 64 * CIN; id += 256) {
    int cl = id / CIN, ci = id % CIN;
    b_lds[cl * CIN + ci] = (__bf16)W[(size_t)(c0 + cl) * CIN + ci];
  }
  __syncthreads();

  const int w  = t >> 5, l = t & 31;
  const int mt = w & 3;          // row tile 0..3
  const int lh = l >> 4;         // lane half
  const int ll = l & 15;

  for (int cp = 0; cp < 2; ++cp) {
    const int ct = (w >> 2) * 2 + cp;   // col tile 0..3
    v8f acc = {};
#pragma unroll
    for (int k0 = 0; k0 < CIN; k0 += 32) {
      v16bf afr = make_afrag(&a_lds[(16 * mt + ll) * CIN + k0 + 16 * lh]);
      v16bf bfr = *(const v16bf*)&b_lds[(16 * ct + ll) * CIN + k0 + 16 * lh];
      acc = __builtin_amdgcn_wmma_f32_16x16x32_bf16(
          false, afr, false, bfr, (short)0, acc, false, false);
    }
#pragma unroll
    for (int r = 0; r < 8; ++r) {
      int c = c0 + 16 * ct + ll;
      int n = n0 + 16 * mt + r + 8 * lh;
      float v = acc[r] + bias[c];
      if (OUT_NC) out[((size_t)b * NN + n) * CC + c] = (__bf16)v;
      else        out[((size_t)b * CC + c) * NN + n] = (__bf16)v;
    }
  }
}

// ===========================================================================
// Flash attention with double-buffered async K/V tile streaming.
// Per (b, 64-row block): 4 waves x 16 rows. Q staged in LDS once.
// K tile (32x256, [m][c]) and V tile (256x32, [c][m]) so every WMMA
// B-fragment is one aligned 32-byte LDS load. Online softmax; O in 16 v8f
// WMMA tiles. Prefetch of tile i+1 overlaps compute of tile i via ASYNCcnt.
// ===========================================================================
__global__ __launch_bounds__(128) void attn_kernel(
    const __bf16* __restrict__ qT,    // [B][N][C]
    const __bf16* __restrict__ kT,    // [B][M][C]
    const __bf16* __restrict__ vCM,   // [B][C][M]
    const float*  __restrict__ dg,    // dgcnn_feat [B][C][N]
    const float*  __restrict__ gamma,
    float* __restrict__ fused,        // [B][C][N]
    float* __restrict__ stats)        // [2*C] (sum, sumsq)
{
  __shared__ __bf16 q_lds[64 * CC];         // [n_local][c]     32 KB
  __shared__ __bf16 k_lds[2][32 * CC];      // [buf][m][c]      32 KB
  __shared__ __bf16 v_lds[2][CC * 32];      // [buf][c][m]      32 KB
  __shared__ float  s_lds[4][16 * 32];      //                   8 KB
  __shared__ __bf16 p_lds[4][16 * 32];      //                   4 KB
  __shared__ float  corr_lds[4][16];
  __shared__ float  stat_lds[4][16];

  const int b    = blockIdx.y;
  const int nblk = blockIdx.x;
  const int t    = threadIdx.x;
  const int w    = t >> 5, l = t & 31;
  const int lh   = l >> 4, ll = l & 15;
  const int row16 = nblk * 64 + w * 16;   // first n-row of this wave

  // 16 async instructions per wave per tile (8 for K + 8 for V)
  auto prefetch_tile = [&](int m0, int buf) {
    for (int id = t; id < 1024; id += 128) {
      int ml = id >> 5, part = id & 31;
      async_copy_b128(kT + ((size_t)b * NN + m0 + ml) * CC + part * 8,
                      &k_lds[buf][ml * CC + part * 8]);
    }
    for (int id = t; id < 1024; id += 128) {
      int c = id >> 2, part = id & 3;
      async_copy_b128(vCM + ((size_t)b * CC + c) * NN + m0 + part * 8,
                      &v_lds[buf][c * 32 + part * 8]);
    }
  };

  // ---- stage Q block + first K/V tile ------------------------------------
  for (int id = t; id < 2048; id += 128) {
    int nl = id >> 5, part = id & 31;
    async_copy_b128(qT + ((size_t)b * NN + nblk * 64 + nl) * CC + part * 8,
                    &q_lds[nl * CC + part * 8]);
  }
  prefetch_tile(0, 0);
  wait_async0();
  __syncthreads();

  v8f o[16] = {};
  float run_m = -3.0e38f;
  float run_s = 0.0f;

  for (int m0 = 0; m0 < NN; m0 += 32) {
    const int buf = (m0 >> 5) & 1;

    // ---- software pipeline: issue next tile, then drain current ----------
    if (m0 + 32 < NN) {
      prefetch_tile(m0 + 32, buf ^ 1);
      wait_async16();               // everything older than next tile done
    } else {
      wait_async0();
    }
    __syncthreads();

    // ---- S = Q * Kt (16 x 32), two 16x16 WMMA tiles, K-loop of 8 ---------
    v8f s0 = {}, s1 = {};
#pragma unroll
    for (int ks = 0; ks < 8; ++ks) {
      v16bf af = make_afrag(&q_lds[(w * 16 + ll) * CC + ks * 32 + 16 * lh]);
      v16bf b0 = *(const v16bf*)&k_lds[buf][(ll) * CC + ks * 32 + 16 * lh];
      v16bf b1 = *(const v16bf*)&k_lds[buf][(16 + ll) * CC + ks * 32 + 16 * lh];
      s0 = __builtin_amdgcn_wmma_f32_16x16x32_bf16(
          false, af, false, b0, (short)0, s0, false, false);
      s1 = __builtin_amdgcn_wmma_f32_16x16x32_bf16(
          false, af, false, b1, (short)0, s1, false, false);
    }
#pragma unroll
    for (int r = 0; r < 8; ++r) {
      s_lds[w][(r + 8 * lh) * 32 + ll]      = s0[r];
      s_lds[w][(r + 8 * lh) * 32 + 16 + ll] = s1[r];
    }
    __syncthreads();

    // ---- online softmax: 2 lanes per row, 16 cols each -------------------
    {
      const int row = ll;
      const float* sr = &s_lds[w][row * 32 + 16 * lh];
      float tmax = -3.0e38f;
#pragma unroll
      for (int i = 0; i < 16; ++i) tmax = fmaxf(tmax, sr[i]);
      tmax = fmaxf(tmax, __shfl_xor(tmax, 16));
      float nm   = fmaxf(run_m, tmax);
      float corr = __expf(run_m - nm);
      float psum = 0.f;
#pragma unroll
      for (int i = 0; i < 16; ++i) {
        float e = __expf(sr[i] - nm);
        psum += e;
        p_lds[w][row * 32 + 16 * lh + i] = (__bf16)e;
      }
      psum += __shfl_xor(psum, 16);
      run_s = run_s * corr + psum;
      run_m = nm;
      if (lh == 0) corr_lds[w][row] = corr;
    }
    __syncthreads();

    // ---- rescale O by per-row correction ---------------------------------
    {
      float f[8];
#pragma unroll
      for (int r = 0; r < 8; ++r) f[r] = corr_lds[w][r + 8 * lh];
#pragma unroll
      for (int ctile = 0; ctile < 16; ++ctile)
#pragma unroll
        for (int r = 0; r < 8; ++r) o[ctile][r] *= f[r];
    }

    // ---- O += P * V  (A = P 16x32, 16 column tiles of C, K = 32) ---------
    {
      v16bf pf = make_afrag(&p_lds[w][ll * 32 + 16 * lh]);
#pragma unroll
      for (int ctile = 0; ctile < 16; ++ctile) {
        v16bf bf = *(const v16bf*)&v_lds[buf][(ctile * 16 + ll) * 32 + 16 * lh];
        o[ctile] = __builtin_amdgcn_wmma_f32_16x16x32_bf16(
            false, pf, false, bf, (short)0, o[ctile], false, false);
      }
    }
    __syncthreads();   // all waves done reading buf before it is re-filled
  }

  // ---- finalize: O / rowsum, residual + gamma, BN stats ------------------
  if (lh == 0) stat_lds[w][ll] = 1.0f / run_s;
  __syncthreads();
  float inv[8];
#pragma unroll
  for (int r = 0; r < 8; ++r) inv[r] = stat_lds[w][r + 8 * lh];

  const float g = gamma[0];
#pragma unroll
  for (int ctile = 0; ctile < 16; ++ctile) {
    int c = ctile * 16 + ll;
    float ssum = 0.f, ssq = 0.f;
#pragma unroll
    for (int r = 0; r < 8; ++r) {
      int n = row16 + r + 8 * lh;
      float att = o[ctile][r] * inv[r];
      float fv  = dg[((size_t)b * CC + c) * NN + n] + g * att;
      fused[((size_t)b * CC + c) * NN + n] = fv;
      ssum += fv; ssq += fv * fv;
    }
    atomicAdd(&stats[c], ssum);
    atomicAdd(&stats[CC + c], ssq);
  }
}

// ===========================================================================
__global__ void zero_stats(float* stats) {
  int i = threadIdx.x + blockIdx.x * blockDim.x;
  if (i < 2 * CC) stats[i] = 0.f;
}

__global__ __launch_bounds__(256) void bn_apply(
    const float* __restrict__ fused, const float* __restrict__ stats,
    const float* __restrict__ bw, const float* __restrict__ bb,
    float* __restrict__ out)
{
  size_t idx = (size_t)blockIdx.x * 256 + threadIdx.x;
  const size_t total = (size_t)BB * CC * NN;
  if (idx >= total) return;
  int c = (int)((idx / NN) % CC);
  const float cnt = (float)(BB * NN);
  float mean = stats[c] / cnt;
  float var  = stats[CC + c] / cnt - mean * mean;
  float v = (fused[idx] - mean) * rsqrtf(var + BN_EPS) * bw[c] + bb[c];
  out[idx] = fmaxf(v, 0.f);
}

// ===========================================================================
extern "C" void kernel_launch(void* const* d_in, const int* in_sizes, int n_in,
                              void* d_out, int out_size, void* d_ws, size_t ws_size,
                              hipStream_t stream) {
  const float* dg    = (const float*)d_in[0];
  const float* geo   = (const float*)d_in[1];
  const float* Wq    = (const float*)d_in[2];
  const float* bq    = (const float*)d_in[3];
  const float* Wk    = (const float*)d_in[4];
  const float* bk    = (const float*)d_in[5];
  const float* Wv    = (const float*)d_in[6];
  const float* bv    = (const float*)d_in[7];
  const float* gam   = (const float*)d_in[8];
  const float* bnw   = (const float*)d_in[9];
  const float* bnb   = (const float*)d_in[10];
  float* out = (float*)d_out;

  char* ws = (char*)d_ws;
  __bf16* qT   = (__bf16*)(ws);                        //  8 MB  [B][N][C]
  __bf16* kT   = (__bf16*)(ws + ((size_t)8  << 20));   //  8 MB  [B][M][C]
  __bf16* vCM  = (__bf16*)(ws + ((size_t)16 << 20));   //  8 MB  [B][C][M]
  float*  fus  = (float*) (ws + ((size_t)24 << 20));   // 16 MB  [B][C][N]
  float*  stats= (float*) (ws + ((size_t)40 << 20));   //  2 KB

  zero_stats<<<2, 256, 0, stream>>>(stats);

  dim3 gproj(NN / 64, CC / 64, BB);
  proj_kernel<C1, true ><<<gproj, 256, 0, stream>>>(dg,  Wq, bq, qT);
  proj_kernel<C2, true ><<<gproj, 256, 0, stream>>>(geo, Wk, bk, kT);
  proj_kernel<C2, false><<<gproj, 256, 0, stream>>>(geo, Wv, bv, vCM);

  attn_kernel<<<dim3(NN / 64, BB), 128, 0, stream>>>(qT, kT, vCM, dg, gam, fus, stats);

  const size_t total = (size_t)BB * CC * NN;
  bn_apply<<<(unsigned)((total + 255) / 256), 256, 0, stream>>>(fus, stats, bnw, bnb, out);
}
